// MSFCN1_CCCA2_71073118814814
// MI455X (gfx1250) — matrix operand order repaired and checked by reading, not compiled, not run on hardware
//
#include <hip/hip_runtime.h>
#include <hip/hip_bf16.h>

typedef __attribute__((ext_vector_type(16))) _Float16 v16h;
typedef __attribute__((ext_vector_type(8)))  float    v8f;

#define B_   16
#define CIN_ 512
#define V_   96
#define T_   96
#define C_   64
#define VT_  (V_ * T_)

#define KCHUNK 64
#define KPAD   72

struct GP {
  const void* A;  const void* Bp;  void* O;
  const float* bias; const float* resid; const float* gammaPtr;
  long long as_b, as_i1, as_i2, as_m, as_k;
  long long bs_b, bs_i1, bs_i2, bs_n, bs_k;
  long long os_b, os_i1, os_i2, os_m, os_n;
  int bias_i1; float bias_scale;
  int M, N, K;
  int ni1, ni2;
  int a_f16, b_f16, o_f16, accum, mask_diag;
};

// Stage a (rows x kc) f16 tile with unit k-stride into LDS via the CDNA5
// async Global->LDS DMA path (4 bytes = one f16 pair per lane per issue).
// Caller guarantees: (base | s_m) even, kc in {32,64}; completion is
// tracked with ASYNCcnt (s_wait_asynccnt before the consuming barrier).
__device__ __forceinline__ void stage_async_f16(const _Float16* gsrc, long long base,
                                                long long s_m, int rows, int kc, int k0,
                                                _Float16* lds, int tid) {
  const int p2   = kc >> 1;                 // f16 pairs per row: 16 or 32
  const int psh  = (kc == 64) ? 5 : 4;
  const int total = rows * p2;
  for (int idx = tid; idx < total; idx += 256) {
    const int m  = idx >> psh;
    const int kp = idx & (p2 - 1);
    const _Float16* g = gsrc + base + (long long)m * s_m + (k0 + 2 * kp);
    unsigned lofs = (unsigned)(size_t)(void*)&lds[m * KPAD + 2 * kp];
    asm volatile("global_load_async_to_lds_b32 %0, %1, off"
                 :: "v"(lofs), "v"(g) : "memory");
  }
}

// Generic batched WMMA GEMM: Out[m,n] = sum_k A[m,k]*B[n,k] with fully
// general element strides; f16 operands staged via async Global->LDS DMA
// when k-contiguous (else converted through VGPRs); 2-deep LDS double
// buffering overlaps the async stage of chunk i+1 with WMMA compute of
// chunk i; f32 accumulate via v_wmma_f32_16x16x32_f16; optional bias /
// diag mask / residual (gamma*y + x) / accumulate-into-out epilogue.
template <int MT, int NT>
__global__ __launch_bounds__(256) void wmma_gemm(GP p) {
  constexpr int Mc = MT * 16;
  constexpr int Nc = NT * 16;
  constexpr int NTILES = MT * NT;
  constexpr int TPW = (NTILES + 7) / 8;     // tiles per wave (compile-time)

  __shared__ _Float16 Alds[2][Mc * KPAD];
  __shared__ _Float16 Blds[2][Nc * KPAD];

  const int bid = blockIdx.x;
  const int i2  = bid % p.ni2;
  const int i1  = (bid / p.ni2) % p.ni1;
  const int b   = bid / (p.ni2 * p.ni1);

  const long long abase = (long long)b * p.as_b + (long long)i1 * p.as_i1 + (long long)i2 * p.as_i2;
  const long long bbase = (long long)b * p.bs_b + (long long)i1 * p.bs_i1 + (long long)i2 * p.bs_i2;
  const long long obase = (long long)b * p.os_b + (long long)i1 * p.os_i1 + (long long)i2 * p.os_i2;

  const int tid  = threadIdx.x;
  const int wave = tid >> 5;
  const int lane = tid & 31;
  const int half = lane >> 4;
  const int l16  = lane & 15;

  const bool a_async = p.a_f16 && (p.as_k == 1) && (((abase | p.as_m) & 1) == 0);
  const bool b_async = p.b_f16 && (p.bs_k == 1) && (((bbase | p.bs_n) & 1) == 0);

  // Stage both operand tiles for K chunk starting at k0 into buffer `buf`.
  auto stage = [&](int buf, int k0, int kc) {
    const int ksh = (kc == 64) ? 6 : 5;
    if (a_async) {
      stage_async_f16((const _Float16*)p.A, abase, p.as_m, Mc, kc, k0, Alds[buf], tid);
    } else if (p.a_f16) {
      const _Float16* Ap = (const _Float16*)p.A;
      for (int idx = tid; idx < Mc * kc; idx += 256) {
        int m = idx >> ksh, k = idx & (kc - 1);
        Alds[buf][m * KPAD + k] = Ap[abase + (long long)m * p.as_m + (long long)(k0 + k) * p.as_k];
      }
    } else {
      const float* Ap = (const float*)p.A;
      for (int idx = tid; idx < Mc * kc; idx += 256) {
        int m = idx >> ksh, k = idx & (kc - 1);
        Alds[buf][m * KPAD + k] = (_Float16)Ap[abase + (long long)m * p.as_m + (long long)(k0 + k) * p.as_k];
      }
    }
    if (b_async) {
      stage_async_f16((const _Float16*)p.Bp, bbase, p.bs_n, Nc, kc, k0, Blds[buf], tid);
    } else if (p.b_f16) {
      const _Float16* Bq = (const _Float16*)p.Bp;
      for (int idx = tid; idx < Nc * kc; idx += 256) {
        int n = idx >> ksh, k = idx & (kc - 1);
        Blds[buf][n * KPAD + k] = Bq[bbase + (long long)n * p.bs_n + (long long)(k0 + k) * p.bs_k];
      }
    } else {
      const float* Bq = (const float*)p.Bp;
      for (int idx = tid; idx < Nc * kc; idx += 256) {
        int n = idx >> ksh, k = idx & (kc - 1);
        Blds[buf][n * KPAD + k] = (_Float16)Bq[bbase + (long long)n * p.bs_n + (long long)(k0 + k) * p.bs_k];
      }
    }
  };

  v8f acc[TPW];
  #pragma unroll
  for (int t = 0; t < TPW; ++t) acc[t] = (v8f){0.f,0.f,0.f,0.f,0.f,0.f,0.f,0.f};

  const int nch = (p.K + KCHUNK - 1) / KCHUNK;
  {
    const int kc0 = (p.K < KCHUNK) ? p.K : KCHUNK;
    stage(0, 0, kc0);
  }

  for (int i = 0; i < nch; ++i) {
    const int cur = i & 1;
    const int k0  = i * KCHUNK;
    const int kc  = (p.K - k0 < KCHUNK) ? (p.K - k0) : KCHUNK;   // 64 or 32

    // Chunk i's async copies (and any older) complete, then all waves align.
    asm volatile("s_wait_asynccnt 0x0" ::: "memory");
    __syncthreads();

    // Overlap: issue the stage of chunk i+1 into the other buffer while the
    // WMMA loop below consumes chunk i. (Barrier at loop end guarantees the
    // target buffer is no longer being read.)
    if (i + 1 < nch) {
      const int k0n = k0 + KCHUNK;
      const int kcn = (p.K - k0n < KCHUNK) ? (p.K - k0n) : KCHUNK;
      stage(cur ^ 1, k0n, kcn);
      // hint the chunk after next into the cache (global_prefetch_b8)
      if (i + 2 < nch && tid < Mc) {
        long long g = abase + (long long)tid * p.as_m + (long long)(k0n + KCHUNK) * p.as_k;
        __builtin_prefetch((const char*)p.A + g * (p.a_f16 ? 2 : 4), 0, 1);
      }
    }

    // ---- WMMA over chunk i: register-resident accumulators ----
    #pragma unroll
    for (int tc = 0; tc < TPW; ++tc) {
      const int t = wave + tc * 8;
      if (t < NTILES) {
        const int mt = t / NT, nt = t - mt * NT;
        for (int kb = 0; kb < kc; kb += 32) {
          union { v16h h; unsigned u[8]; } av, bv;
          const _Float16* arow = &Alds[cur][(mt * 16 + l16) * KPAD + kb];
          const _Float16* brow = &Blds[cur][(nt * 16 + l16) * KPAD + kb];
          #pragma unroll
          for (int j = 0; j < 8; ++j) {
            // ISA 16-bit A 16x32 layout: VGPR j holds K pair at
            // ((j&3)*2 + (j>>2)*16 + half*8); both lane halves M=l16.
            int ka = ((j & 3) << 1) + ((j >> 2) << 4) + (half << 3);
            av.u[j] = *reinterpret_cast<const unsigned*>(arow + ka);
            // B 32x16 layout: VGPR j holds K pair at
            // ((j&3)*2 + (j>>2)*16 + half*8), N = l16 (same as A).
            int kbd = ((j & 3) << 1) + ((j >> 2) << 4) + (half << 3);
            bv.u[j] = *reinterpret_cast<const unsigned*>(brow + kbd);
          }
          acc[tc] = __builtin_amdgcn_wmma_f32_16x16x32_f16(
              false, av.h, false, bv.h, (short)0, acc[tc], false, false);
        }
      }
    }
    __syncthreads();
  }

  // ---- epilogue ----
  const float gval = p.resid ? p.gammaPtr[0] : 0.f;
  #pragma unroll
  for (int tc = 0; tc < TPW; ++tc) {
    const int t = wave + tc * 8;
    if (t < NTILES) {
      const int mt = t / NT, nt = t - mt * NT;
      const int gn = nt * 16 + l16;
      #pragma unroll
      for (int r = 0; r < 8; ++r) {
        const int gm = mt * 16 + (half << 3) + r;   // C/D layout: M = half*8 + r
        float v = acc[tc][r];
        if (p.bias) v += p.bias_scale * p.bias[p.bias_i1 * i1 + gm];
        if (p.mask_diag && gm == gn) v = -1e30f;
        const long long oi = obase + (long long)gm * p.os_m + (long long)gn * p.os_n;
        if (p.resid) v = gval * v + p.resid[oi];
        if (p.o_f16) {
          ((_Float16*)p.O)[oi] = (_Float16)v;
        } else {
          float* op = (float*)p.O;
          op[oi] = p.accum ? (op[oi] + v) : v;
        }
      }
    }
  }
}

// Joint softmax over two concatenated score parts; one wave per row.
__global__ __launch_bounds__(256) void softmax_rows(const float* s1, int L1,
                                                    const float* s2, int L2,
                                                    _Float16* a1, _Float16* a2,
                                                    int nrows) {
  const int row  = blockIdx.x * 8 + (threadIdx.x >> 5);
  const int lane = threadIdx.x & 31;
  if (row >= nrows) return;
  const float* r1 = s1 + (long long)row * L1;
  const float* r2 = s2 + (long long)row * L2;
  const int n1 = L1 >> 5, n2 = L2 >> 5;
  float e[6];
  float mx = -3.4e38f;
  for (int i = 0; i < n1; ++i) { e[i]      = r1[lane + (i << 5)]; mx = fmaxf(mx, e[i]); }
  for (int i = 0; i < n2; ++i) { e[n1 + i] = r2[lane + (i << 5)]; mx = fmaxf(mx, e[n1 + i]); }
  for (int s = 16; s > 0; s >>= 1) mx = fmaxf(mx, __shfl_xor(mx, s, 32));
  float sum = 0.f;
  for (int i = 0; i < n1 + n2; ++i) { e[i] = __expf(e[i] - mx); sum += e[i]; }
  for (int s = 16; s > 0; s >>= 1) sum += __shfl_xor(sum, s, 32);
  const float inv = 1.f / sum;
  for (int i = 0; i < n1; ++i) a1[(long long)row * L1 + lane + (i << 5)] = (_Float16)(e[i] * inv);
  for (int i = 0; i < n2; ++i) a2[(long long)row * L2 + lane + (i << 5)] = (_Float16)(e[n1 + i] * inv);
}

__global__ void zero_f32(float* p, long long n) {
  long long i = (long long)blockIdx.x * 256 + threadIdx.x;
  const long long st = (long long)gridDim.x * 256;
  for (; i < n; i += st) p[i] = 0.f;
}

static GP gp_default() {
  GP p;
  p.A = nullptr; p.Bp = nullptr; p.O = nullptr;
  p.bias = nullptr; p.resid = nullptr; p.gammaPtr = nullptr;
  p.as_b = p.as_i1 = p.as_i2 = p.as_m = p.as_k = 0;
  p.bs_b = p.bs_i1 = p.bs_i2 = p.bs_n = p.bs_k = 0;
  p.os_b = p.os_i1 = p.os_i2 = p.os_m = p.os_n = 0;
  p.bias_i1 = 0; p.bias_scale = 1.f;
  p.M = p.N = p.K = 16; p.ni1 = 1; p.ni2 = 1;
  p.a_f16 = p.b_f16 = 1; p.o_f16 = 0; p.accum = 0; p.mask_diag = 0;
  return p;
}

static void launch_gemm(const GP& p, hipStream_t stream) {
  const int grid = B_ * p.ni1 * p.ni2;
  const int MT = p.M >> 4, NT = p.N >> 4;
  if      (MT == 4 && NT == 8) wmma_gemm<4, 8><<<grid, 256, 0, stream>>>(p);
  else if (MT == 4 && NT == 4) wmma_gemm<4, 4><<<grid, 256, 0, stream>>>(p);
  else if (MT == 6 && NT == 6) wmma_gemm<6, 6><<<grid, 256, 0, stream>>>(p);
  else if (MT == 4 && NT == 6) wmma_gemm<4, 6><<<grid, 256, 0, stream>>>(p);
}

extern "C" void kernel_launch(void* const* d_in, const int* in_sizes, int n_in,
                              void* d_out, int out_size, void* d_ws, size_t ws_size,
                              hipStream_t stream) {
  (void)in_sizes; (void)n_in; (void)out_size; (void)ws_size;
  const float* x     = (const float*)d_in[0];
  const float* Wq    = (const float*)d_in[1];
  const float* bq    = (const float*)d_in[2];
  const float* Wk    = (const float*)d_in[3];
  const float* bk    = (const float*)d_in[4];
  const float* Wv    = (const float*)d_in[5];
  const float* bv    = (const float*)d_in[6];
  const float* Wo    = (const float*)d_in[7];
  const float* bo    = (const float*)d_in[8];
  const float* gamma = (const float*)d_in[9];
  float* out = (float*)d_out;

  const long long BCVT = (long long)B_ * C_ * VT_;          // 9,437,184
  const long long SMAX = (long long)B_ * V_ * T_ * V_;      // 14,155,776

  _Float16* qh    = (_Float16*)d_ws;
  _Float16* kh    = qh + BCVT;
  _Float16* vh    = kh + BCVT;
  float*    outws = (float*)(vh + BCVT);
  float*    S1    = outws + BCVT;
  float*    S2    = S1 + SMAX;
  _Float16* A1h   = (_Float16*)(S2 + SMAX);
  _Float16* A2h   = A1h + SMAX;

  // 0) zero the output accumulator
  zero_f32<<<4096, 256, 0, stream>>>(outws, BCVT);

  // 1) QKV projections: q/k/v[b,c,vt] = W(64x512) . x[b,:,vt] + bias (f16 out)
  {
    const float* Ws[3] = {Wq, Wk, Wv};
    const float* bs[3] = {bq, bk, bv};
    _Float16*    Os[3] = {qh, kh, vh};
    for (int i = 0; i < 3; ++i) {
      GP p = gp_default();
      p.A = Ws[i]; p.a_f16 = 0; p.as_m = CIN_; p.as_k = 1;
      p.Bp = x;    p.b_f16 = 0; p.bs_b = (long long)CIN_ * VT_; p.bs_i2 = 128; p.bs_n = 1; p.bs_k = VT_;
      p.O = Os[i]; p.o_f16 = 1; p.os_b = (long long)C_ * VT_;   p.os_i2 = 128; p.os_m = VT_; p.os_n = 1;
      p.bias = bs[i]; p.bias_scale = 1.f;
      p.M = 64; p.N = 128; p.K = CIN_; p.ni1 = 1; p.ni2 = VT_ / 128;   // 72
      launch_gemm(p, stream);
    }
  }

  // ---------------- Family A: rows (b,c,v); parts e_vcc (64) + e_cvv (96) ----
  { // e_vcc[b,c,v,d] = sum_t q[c,v,t] k[d,v,t], diag(c==d) masked
    GP p = gp_default();
    p.A = qh; p.as_b = (long long)C_ * VT_; p.as_i2 = T_; p.as_m = VT_; p.as_k = 1;
    p.Bp = kh; p.bs_b = (long long)C_ * VT_; p.bs_i2 = T_; p.bs_n = VT_; p.bs_k = 1;
    p.O = S1; p.os_b = (long long)C_ * V_ * C_; p.os_i2 = C_; p.os_m = (long long)V_ * C_; p.os_n = 1;
    p.M = 64; p.N = 64; p.K = T_; p.ni2 = V_; p.mask_diag = 1;
    launch_gemm(p, stream);
  }
  { // e_cvv[b,c,v,w] = sum_t q[c,v,t] k[c,w,t]
    GP p = gp_default();
    p.A = qh; p.as_b = (long long)C_ * VT_; p.as_i2 = VT_; p.as_m = T_; p.as_k = 1;
    p.Bp = kh; p.bs_b = (long long)C_ * VT_; p.bs_i2 = VT_; p.bs_n = T_; p.bs_k = 1;
    p.O = S2; p.os_b = (long long)C_ * V_ * V_; p.os_i2 = (long long)V_ * V_; p.os_m = V_; p.os_n = 1;
    p.M = V_; p.N = V_; p.K = T_; p.ni2 = C_;
    launch_gemm(p, stream);
  }
  softmax_rows<<<(B_ * C_ * V_) / 8, 256, 0, stream>>>(S1, C_, S2, V_, A1h, A2h, B_ * C_ * V_);
  { // out[b,d,v,t] += sum_c attn_vcc[b,c,v,d] * v[b,c,v,t]
    GP p = gp_default();
    p.A = A1h; p.as_b = (long long)C_ * V_ * C_; p.as_i2 = C_; p.as_m = 1; p.as_k = (long long)V_ * C_;
    p.Bp = vh; p.bs_b = (long long)C_ * VT_; p.bs_i2 = T_; p.bs_n = 1; p.bs_k = VT_;
    p.O = outws; p.os_b = (long long)C_ * VT_; p.os_i2 = T_; p.os_m = VT_; p.os_n = 1;
    p.M = 64; p.N = T_; p.K = 64; p.ni2 = V_; p.accum = 1;
    launch_gemm(p, stream);
  }
  { // out[b,c,w,t] += sum_v attn_cvv[b,c,v,w] * v[b,c,v,t]
    GP p = gp_default();
    p.A = A2h; p.as_b = (long long)C_ * V_ * V_; p.as_i2 = (long long)V_ * V_; p.as_m = 1; p.as_k = V_;
    p.Bp = vh; p.bs_b = (long long)C_ * VT_; p.bs_i2 = VT_; p.bs_n = 1; p.bs_k = T_;
    p.O = outws; p.os_b = (long long)C_ * VT_; p.os_i2 = VT_; p.os_m = T_; p.os_n = 1;
    p.M = V_; p.N = T_; p.K = V_; p.ni2 = C_; p.accum = 1;
    launch_gemm(p, stream);
  }

  // ---------------- Family B: rows (b,c,t); parts e_tcc (64) + e_ctt (96) ----
  { // e_tcc[b,c,t,d] = sum_v q[c,v,t] k[d,v,t], diag masked
    GP p = gp_default();
    p.A = qh; p.as_b = (long long)C_ * VT_; p.as_i2 = 1; p.as_m = VT_; p.as_k = T_;
    p.Bp = kh; p.bs_b = (long long)C_ * VT_; p.bs_i2 = 1; p.bs_n = VT_; p.bs_k = T_;
    p.O = S1; p.os_b = (long long)C_ * T_ * C_; p.os_i2 = C_; p.os_m = (long long)T_ * C_; p.os_n = 1;
    p.M = 64; p.N = 64; p.K = V_; p.ni2 = T_; p.mask_diag = 1;
    launch_gemm(p, stream);
  }
  { // e_ctt[b,c,t,s] = sum_v q[c,v,t] k[c,v,s]
    GP p = gp_default();
    p.A = qh; p.as_b = (long long)C_ * VT_; p.as_i2 = VT_; p.as_m = 1; p.as_k = T_;
    p.Bp = kh; p.bs_b = (long long)C_ * VT_; p.bs_i2 = VT_; p.bs_n = 1; p.bs_k = T_;
    p.O = S2; p.os_b = (long long)C_ * T_ * T_; p.os_i2 = (long long)T_ * T_; p.os_m = T_; p.os_n = 1;
    p.M = T_; p.N = T_; p.K = V_; p.ni2 = C_;
    launch_gemm(p, stream);
  }
  softmax_rows<<<(B_ * C_ * T_) / 8, 256, 0, stream>>>(S1, C_, S2, T_, A1h, A2h, B_ * C_ * T_);
  { // out[b,d,v,t] += sum_c attn_tcc[b,c,t,d] * v[b,c,v,t]
    GP p = gp_default();
    p.A = A1h; p.as_b = (long long)C_ * T_ * C_; p.as_i2 = C_; p.as_m = 1; p.as_k = (long long)T_ * C_;
    p.Bp = vh; p.bs_b = (long long)C_ * VT_; p.bs_i2 = 1; p.bs_n = T_; p.bs_k = VT_;
    p.O = outws; p.os_b = (long long)C_ * VT_; p.os_i2 = 1; p.os_m = VT_; p.os_n = T_;
    p.M = 64; p.N = V_; p.K = 64; p.ni2 = T_; p.accum = 1;
    launch_gemm(p, stream);
  }
  { // out[b,c,v,s] += sum_t v[b,c,v,t] * attn_ctt[b,c,t,s]
    GP p = gp_default();
    p.A = vh; p.as_b = (long long)C_ * VT_; p.as_i2 = VT_; p.as_m = T_; p.as_k = 1;
    p.Bp = A2h; p.bs_b = (long long)C_ * T_ * T_; p.bs_i2 = (long long)T_ * T_; p.bs_n = 1; p.bs_k = T_;
    p.O = outws; p.os_b = (long long)C_ * VT_; p.os_i2 = VT_; p.os_m = T_; p.os_n = 1;
    p.M = V_; p.N = T_; p.K = T_; p.ni2 = C_; p.accum = 1;
    launch_gemm(p, stream);
  }

  // ---------------- Family C: rows (b,v,t); parts e_tvv (96) + e_vtt (96) ----
  { // e_tvv[b,v,t,w] = sum_c q[c,v,t] k[c,w,t], diag(v==w) masked
    GP p = gp_default();
    p.A = qh; p.as_b = (long long)C_ * VT_; p.as_i2 = 1; p.as_m = T_; p.as_k = VT_;
    p.Bp = kh; p.bs_b = (long long)C_ * VT_; p.bs_i2 = 1; p.bs_n = T_; p.bs_k = VT_;
    p.O = S1; p.os_b = (long long)V_ * T_ * V_; p.os_i2 = V_; p.os_m = (long long)T_ * V_; p.os_n = 1;
    p.M = V_; p.N = V_; p.K = C_; p.ni2 = T_; p.mask_diag = 1;
    launch_gemm(p, stream);
  }
  { // e_vtt[b,v,t,s] = sum_c q[c,v,t] k[c,v,s]
    GP p = gp_default();
    p.A = qh; p.as_b = (long long)C_ * VT_; p.as_i2 = T_; p.as_m = 1; p.as_k = VT_;
    p.Bp = kh; p.bs_b = (long long)C_ * VT_; p.bs_i2 = T_; p.bs_n = 1; p.bs_k = VT_;
    p.O = S2; p.os_b = (long long)V_ * T_ * T_; p.os_i2 = (long long)T_ * T_; p.os_m = T_; p.os_n = 1;
    p.M = T_; p.N = T_; p.K = C_; p.ni2 = V_;
    launch_gemm(p, stream);
  }
  softmax_rows<<<(B_ * V_ * T_) / 8, 256, 0, stream>>>(S1, V_, S2, T_, A1h, A2h, B_ * V_ * T_);
  { // out[b,c,w,t] += sum_v v[b,c,v,t] * attn_tvv[b,v,t,w]
    GP p = gp_default();
    p.A = vh; p.as_b = (long long)C_ * VT_; p.as_i2 = 1; p.as_m = VT_; p.as_k = T_;
    p.Bp = A1h; p.bs_b = (long long)V_ * T_ * V_; p.bs_i2 = V_; p.bs_n = 1; p.bs_k = (long long)T_ * V_;
    p.O = outws; p.os_b = (long long)C_ * VT_; p.os_i2 = 1; p.os_m = VT_; p.os_n = T_;
    p.M = 64; p.N = V_; p.K = V_; p.ni2 = T_; p.accum = 1;
    launch_gemm(p, stream);
  }
  { // out[b,c,v,s] += sum_t v[b,c,v,t] * attn_vtt[b,v,t,s]
    GP p = gp_default();
    p.A = vh; p.as_b = (long long)C_ * VT_; p.as_i2 = T_; p.as_m = VT_; p.as_k = 1;
    p.Bp = A2h; p.bs_b = (long long)V_ * T_ * T_; p.bs_i2 = (long long)T_ * T_; p.bs_n = 1; p.bs_k = T_;
    p.O = outws; p.os_b = (long long)C_ * VT_; p.os_i2 = T_; p.os_m = VT_; p.os_n = 1;
    p.M = 64; p.N = T_; p.K = T_; p.ni2 = V_; p.accum = 1;
    launch_gemm(p, stream);
  }

  // 3) Final projection + residual: d_out = gamma*(Wo.out + 6*bo) + x
  {
    GP p = gp_default();
    p.A = Wo; p.a_f16 = 0; p.as_m = C_; p.as_k = 1; p.as_i1 = (long long)64 * C_;
    p.Bp = outws; p.b_f16 = 0; p.bs_b = (long long)C_ * VT_; p.bs_i2 = 128; p.bs_n = 1; p.bs_k = VT_;
    p.O = out; p.os_b = (long long)CIN_ * VT_; p.os_i1 = (long long)64 * VT_; p.os_i2 = 128;
    p.os_m = VT_; p.os_n = 1;
    p.bias = bo; p.bias_i1 = 64; p.bias_scale = 6.f;
    p.resid = x; p.gammaPtr = gamma;
    p.M = 64; p.N = 128; p.K = C_; p.ni1 = CIN_ / 64; p.ni2 = VT_ / 128;  // 8 x 72
    launch_gemm(p, stream);
  }
}
